// FLDEncoder_16595753631925
// MI455X (gfx1250) — matrix operand order repaired and checked by reading, not compiled, not run on hardware
//
#include <hip/hip_runtime.h>
#include <math.h>

// ---------------------------------------------------------------------------
// FLD encoder for MI455X (gfx1250, wave32, WMMA).
//   conv(k=51,"same") == GEMM: out[o,t] = sum_{ck} W[o,ck] * P[c][t+k]
//   K ordering: ck = c*52 + k  (k padded 51->52 so every kstep boundary and
//   row-crossing point is even -> B fragments load as 8 aligned ds_load_b32
//   from a dual-parity LDS copy of the padded activation plane).
//   Wave = n-tile, loops over m-tiles; 2-deep software pipeline on k-steps.
//   BatchNorm over (B,T) -> split kernels with global f32 atomic reductions.
// ---------------------------------------------------------------------------

typedef __attribute__((ext_vector_type(16))) __bf16 v16bf;
typedef __attribute__((ext_vector_type(8)))  float  v8f;

#define BATCH 2048
#define TLEN  51
#define PADW  25
#define PROW  120   // padded row: index j = t + k, t<=63, k<=51 -> <=114; 120-52=68 even

__device__ __forceinline__ unsigned short f2bf(float f) {
    union { float f; unsigned int u; } cv; cv.f = f;
    unsigned int u = cv.u;
    u += 0x7FFFu + ((u >> 16) & 1u);            // round-to-nearest-even
    return (unsigned short)(u >> 16);
}

union BFrag { unsigned int d[8]; v16bf v; };

// --------------------------- stats zeroing ---------------------------------
__global__ void zero_stats(float* p, int n) {
    int i = blockIdx.x * blockDim.x + threadIdx.x;
    if (i < n) p[i] = 0.0f;
}

// ------------------- weight -> WMMA A-fragment swizzle ---------------------
// A-fragment 16x32 bf16 layout used here (contiguous K, same as B):
//   lanes 0-15 : M = lane,    halves 0-15 -> K=h     (0..15)
//   lanes 16-31: M = lane-16, halves 0-15 -> K=h+16  (16..31)
// K ordering ck = c*52 + k (k==51 is a zero pad tap).
// frag layout: [kstep][mtile][lane][16 halves]  (32B per lane, coalesced)
__global__ void prep_frags(const float* __restrict__ w, unsigned short* __restrict__ frag,
                           int mtiles, int cin, int ksteps) {
    int idx = blockIdx.x * blockDim.x + threadIdx.x;
    int total = ksteps * mtiles * 32 * 16;
    if (idx >= total) return;
    int h    = idx & 15;
    int lane = (idx >> 4) & 31;
    int mt   = (idx >> 9) % mtiles;
    int ks   = (idx >> 9) / mtiles;
    int kl   = (lane < 16) ? h : h + 16;
    int ck   = ks * 32 + kl;
    int c    = ck / 52;
    int k    = ck - c * 52;
    int o    = mt * 16 + (lane & 15);
    float v  = (k < 51) ? w[((size_t)o * cin + c) * 51 + k] : 0.0f;
    frag[idx] = f2bf(v);
}

// ----------------------------- conv GEMM -----------------------------------
// One workgroup per batch element, 4 waves. Wave -> n-tile.
template<int CIN, int COUT, bool FIRST>
__global__ __launch_bounds__(128) void conv_gemm(
    const float* __restrict__ in, const unsigned short* __restrict__ wfrag,
    const float* __restrict__ bias,
    const float* __restrict__ bnscale, const float* __restrict__ bnshift,
    float* __restrict__ hout, float* __restrict__ gsum, float* __restrict__ gsq)
{
    constexpr int MT     = COUT / 16;
    constexpr int KSTEPS = CIN * 52 / 32;        // 78 or 104 (even)
    constexpr int HW     = CIN * PROW / 2;       // words per parity copy

    // Q[0][x] = halves(2x,2x+1) of P ; Q[1][x] = halves(2x+1,2x+2) of P
    __shared__ unsigned int Qsh[2 * HW];
    __shared__ float red[2 * COUT];

    const int b    = blockIdx.x;
    const int tid  = threadIdx.x;
    const int wave = tid >> 5;
    const int lane = tid & 31;
    const int lhi  = (lane >> 4) & 1;   // B layout: lanes16-31 hold K+16
    const int lml  = lane & 15;

    // init: zero both parity copies + stat partials
    for (int i = tid; i < 2 * HW; i += 128) Qsh[i] = 0u;
    for (int i = tid; i < 2 * COUT; i += 128) red[i] = 0.0f;
    __syncthreads();

    // stage activations (x raw, or BN+ELU(h_prev)) as bf16 into both copies
    unsigned short* Sh = (unsigned short*)Qsh;
    for (int i = tid; i < CIN * TLEN; i += 128) {
        int c = i / TLEN, t = i - c * TLEN;
        float v;
        if (FIRST) {
            v = in[((size_t)b * CIN + c) * TLEN + t];
        } else {
            v = in[((size_t)b * CIN + c) * 64 + t];
            v = v * bnscale[c] + bnshift[c];
            v = (v > 0.0f) ? v : (__expf(v) - 1.0f);
        }
        unsigned short bv = f2bf(v);
        int j = c * PROW + PADW + t;             // j >= 25 > 0
        Sh[j] = bv;                              // parity-0 copy
        Sh[CIN * PROW + j - 1] = bv;             // parity-1 copy (shifted by 1 half)
    }
    __syncthreads();

    const int n = wave * 16 + lml;               // output position t (padded to 64)
    const unsigned int* Qw = (const unsigned int*)Qsh;

    v8f acc[MT];
#pragma unroll
    for (int i = 0; i < MT; ++i)
#pragma unroll
        for (int j = 0; j < 8; ++j) acc[i][j] = 0.0f;

    // fragment loaders ------------------------------------------------------
    auto loadA = [&](int ks, v16bf* af) {
        const unsigned short* abase = wfrag + ((size_t)ks * MT * 32 + lane) * 16;
#pragma unroll
        for (int mt = 0; mt < MT; ++mt)
            af[mt] = *reinterpret_cast<const v16bf*>(abase + (size_t)mt * 512);
    };
    auto gatherB = [&](int ks, BFrag& bf) {
        const int ckb = ks * 32 + lhi * 16;      // even
        const int c0  = (ckb * 5042) >> 18;      // exact ckb/52 for ckb<6553
        const int k0  = ckb - c0 * 52;           // even
        const int kk  = 52 - k0;                 // even; halves >= kk in row c0+1
        const int a0  = c0 * PROW + k0 + n;      // half index of fragment start
        const int wb  = (a0 & 1) * HW + (a0 >> 1);
#pragma unroll
        for (int i = 0; i < 8; ++i)
            bf.d[i] = Qw[wb + i + ((2 * i >= kk) ? 34 : 0)];   // 34 = (PROW-52)/2
    };

    // 2-deep software-pipelined K loop --------------------------------------
    v16bf afA[MT], afB[MT];
    BFrag bfA, bfB;
    loadA(0, afA);
    gatherB(0, bfA);
    for (int ks = 0; ks < KSTEPS; ks += 2) {
        loadA(ks + 1, afB);
        gatherB(ks + 1, bfB);
#pragma unroll
        for (int mt = 0; mt < MT; ++mt)
            acc[mt] = __builtin_amdgcn_wmma_f32_16x16x32_bf16(
                false, afA[mt], false, bfA.v, (short)0, acc[mt], false, false);
        if (ks + 2 < KSTEPS) {
            loadA(ks + 2, afA);
            gatherB(ks + 2, bfA);
        }
#pragma unroll
        for (int mt = 0; mt < MT; ++mt)
            acc[mt] = __builtin_amdgcn_wmma_f32_16x16x32_bf16(
                false, afB[mt], false, bfB.v, (short)0, acc[mt], false, false);
    }

    // epilogue: + bias, store h (stride 64, coalesced in n), per-channel stats
#pragma unroll
    for (int mt = 0; mt < MT; ++mt) {
#pragma unroll
        for (int v = 0; v < 8; ++v) {
            const int o = mt * 16 + lhi * 8 + v; // C/D layout: M = 8*lhi + vgpr
            float val = acc[mt][v] + bias[o];
            if (n < TLEN) {
                hout[((size_t)b * COUT + o) * 64 + n] = val;
                atomicAdd(&red[o], val);
                atomicAdd(&red[COUT + o], val * val);
            }
        }
    }
    __syncthreads();
    for (int o = tid; o < COUT; o += 128) {
        atomicAdd(gsum + o, red[o]);
        atomicAdd(gsq + o, red[COUT + o]);
    }
}

// --------------------------- BN finalize -----------------------------------
__global__ void bn_finalize(const float* __restrict__ sum, const float* __restrict__ sq,
                            const float* __restrict__ g, const float* __restrict__ beta,
                            float* __restrict__ scale, float* __restrict__ shift,
                            int C, float invN) {
    int o = threadIdx.x;
    if (o >= C) return;
    float m   = sum[o] * invN;
    float var = sq[o] * invN - m * m;
    float sc  = g[o] * rsqrtf(var + 1e-5f);
    scale[o] = sc;
    shift[o] = beta[o] - m * sc;
}

// -------------------- spectral + phase-z kernel (per b) ---------------------
__global__ __launch_bounds__(128) void spectral(
    const float* __restrict__ h3, const float* __restrict__ sc3, const float* __restrict__ sh3,
    const float* __restrict__ pw, const float* __restrict__ pb,
    float* __restrict__ o_lat, float* __restrict__ o_freq, float* __restrict__ o_amp,
    float* __restrict__ o_off, float* __restrict__ zbuf,
    float* __restrict__ zsum, float* __restrict__ zsq)
{
    __shared__ float lat[16 * TLEN];
    __shared__ float p2[16], fp[16];
    const int b = blockIdx.x, tid = threadIdx.x;

    if (tid < 16) { p2[tid] = 0.0f; fp[tid] = 0.0f; }
    for (int i = tid; i < 16 * TLEN; i += 128) {
        int c = i / TLEN, t = i - c * TLEN;
        float v = h3[((size_t)b * 16 + c) * 64 + t];
        v = v * sc3[c] + sh3[c];
        v = (v > 0.0f) ? v : (__expf(v) - 1.0f);
        lat[i] = v;
        o_lat[(size_t)b * (16 * TLEN) + i] = v;   // latent output
    }
    __syncthreads();

    // DFT bins f = 0..25 for each of 16 channels
    const float w0 = 6.283185307179586f / 51.0f;
    for (int it = tid; it < 16 * 26; it += 128) {
        int c = it / 26, f = it - c * 26;
        float re = 0.0f, im = 0.0f;
        for (int t = 0; t < TLEN; ++t) {
            float s, co;
            __sincosf(w0 * (float)(f * t), &s, &co);
            float v = lat[c * TLEN + t];
            re += v * co;
            im -= v * s;
        }
        if (f == 0) {
            o_off[b * 16 + c] = re * (1.0f / 51.0f);
        } else {
            float p = re * re + im * im;
            atomicAdd(&p2[c], p);
            atomicAdd(&fp[c], (float)f * p);
        }
    }
    __syncthreads();
    if (tid < 16) {
        float s = p2[tid];
        o_freq[b * 16 + tid] = fp[tid] / s;
        o_amp[b * 16 + tid]  = 2.0f * sqrtf(s) * (1.0f / 51.0f);
    }
    // phase head: z[b,c,o] = <lat[c,:], pw[c,o,:]> + pb[c,o]; batch stats via atomics
    if (tid < 32) {
        int j = tid;                       // j = c*2 + o
        const float* wv = pw + (size_t)j * TLEN;
        float z = pb[j];
        int c = j >> 1;
        for (int t = 0; t < TLEN; ++t) z += lat[c * TLEN + t] * wv[t];
        zbuf[(size_t)b * 32 + j] = z;
        atomicAdd(&zsum[j], z);
        atomicAdd(&zsq[j], z * z);
    }
}

// --------------------------- phase finalize ---------------------------------
__global__ void phase_k(const float* __restrict__ zbuf, const float* __restrict__ zsum,
                        const float* __restrict__ zsq, const float* __restrict__ pg,
                        const float* __restrict__ pbeta, float* __restrict__ o_phase)
{
    int idx = blockIdx.x * blockDim.x + threadIdx.x;   // b*16 + c
    if (idx >= BATCH * 16) return;
    int b = idx >> 4, c = idx & 15;
    float zn[2];
#pragma unroll
    for (int o = 0; o < 2; ++o) {
        int j = c * 2 + o;
        float m   = zsum[j] * (1.0f / (float)BATCH);
        float var = zsq[j] * (1.0f / (float)BATCH) - m * m;
        float z   = zbuf[(size_t)b * 32 + j];
        zn[o] = (z - m) * rsqrtf(var + 1e-5f) * pg[j] + pbeta[j];
    }
    o_phase[idx] = atan2f(zn[1], zn[0]) * 0.15915494309189535f;  // 1/(2*pi)
}

// ----------------------------- host side ------------------------------------
extern "C" void kernel_launch(void* const* d_in, const int* in_sizes, int n_in,
                              void* d_out, int out_size, void* d_ws, size_t ws_size,
                              hipStream_t stream) {
    (void)in_sizes; (void)n_in; (void)out_size; (void)ws_size;

    const float* x   = (const float*)d_in[0];
    const float* w0  = (const float*)d_in[1];
    const float* b0  = (const float*)d_in[2];
    const float* g0  = (const float*)d_in[3];
    const float* be0 = (const float*)d_in[4];
    const float* w1  = (const float*)d_in[5];
    const float* b1  = (const float*)d_in[6];
    const float* g1  = (const float*)d_in[7];
    const float* be1 = (const float*)d_in[8];
    const float* w2  = (const float*)d_in[9];
    const float* b2  = (const float*)d_in[10];
    const float* g2  = (const float*)d_in[11];
    const float* be2 = (const float*)d_in[12];
    const float* pw  = (const float*)d_in[13];
    const float* pb  = (const float*)d_in[14];
    const float* pg  = (const float*)d_in[15];
    const float* pbt = (const float*)d_in[16];

    // workspace layout (floats)
    float* ws   = (float*)d_ws;
    float* h1   = ws;
    float* h2   = h1 + (size_t)BATCH * 64 * 64;
    float* h3   = h2 + (size_t)BATCH * 64 * 64;
    float* zbuf = h3 + (size_t)BATCH * 16 * 64;
    float* st   = zbuf + (size_t)BATCH * 32;
    float *sum1 = st,        *sq1 = st + 64;
    float *sum2 = st + 128,  *sq2 = st + 192;
    float *sum3 = st + 256,  *sq3 = st + 272;
    float *zsum = st + 288,  *zsq = st + 320;       // stats total = 352
    float* sc   = st + 352;
    float *s1 = sc,       *t1 = sc + 64;
    float *s2 = sc + 128, *t2 = sc + 192;
    float *s3 = sc + 256, *t3 = sc + 272;           // scale/shift total = 288
    unsigned short* frag1 = (unsigned short*)(sc + 288);
    unsigned short* frag2 = frag1 + (size_t)78  * 4 * 32 * 16;   // 159744 halves
    unsigned short* frag3 = frag2 + (size_t)104 * 4 * 32 * 16;   // 212992 halves

    // outputs: latent, phase, frequency, amplitude, offset
    float* out     = (float*)d_out;
    float* o_lat   = out;
    float* o_phase = o_lat + (size_t)BATCH * 16 * TLEN;
    float* o_freq  = o_phase + BATCH * 16;
    float* o_amp   = o_freq + BATCH * 16;
    float* o_off   = o_amp + BATCH * 16;

    const float invN = 1.0f / ((float)BATCH * (float)TLEN);

    zero_stats<<<1, 512, 0, stream>>>(st, 352);

    prep_frags<<<(78 * 4 * 512 + 255) / 256, 256, 0, stream>>>(w0, frag1, 4, 48, 78);
    prep_frags<<<(104 * 4 * 512 + 255) / 256, 256, 0, stream>>>(w1, frag2, 4, 64, 104);
    prep_frags<<<(104 * 1 * 512 + 255) / 256, 256, 0, stream>>>(w2, frag3, 1, 64, 104);

    conv_gemm<48, 64, true><<<BATCH, 128, 0, stream>>>(
        x, frag1, b0, nullptr, nullptr, h1, sum1, sq1);
    bn_finalize<<<1, 64, 0, stream>>>(sum1, sq1, g0, be0, s1, t1, 64, invN);

    conv_gemm<64, 64, false><<<BATCH, 128, 0, stream>>>(
        h1, frag2, b1, s1, t1, h2, sum2, sq2);
    bn_finalize<<<1, 64, 0, stream>>>(sum2, sq2, g1, be1, s2, t2, 64, invN);

    conv_gemm<64, 16, false><<<BATCH, 128, 0, stream>>>(
        h2, frag3, b2, s2, t2, h3, sum3, sq3);
    bn_finalize<<<1, 64, 0, stream>>>(sum3, sq3, g2, be2, s3, t3, 16, invN);

    spectral<<<BATCH, 128, 0, stream>>>(h3, s3, t3, pw, pb,
                                        o_lat, o_freq, o_amp, o_off,
                                        zbuf, zsum, zsq);
    phase_k<<<(BATCH * 16 + 255) / 256, 256, 0, stream>>>(zbuf, zsum, zsq, pg, pbt, o_phase);
}
